// GRUPolicy_75453985456271
// MI455X (gfx1250) — matrix operand order repaired and checked by reading, not compiled, not run on hardware
//
#include <hip/hip_runtime.h>
#include <math.h>

typedef _Float16 h16 __attribute__((ext_vector_type(16)));
typedef _Float16 h8v __attribute__((ext_vector_type(8)));
typedef float    f8v __attribute__((ext_vector_type(8)));
typedef unsigned int u4v __attribute__((ext_vector_type(4)));

static __device__ __forceinline__ h16 cat16(h8v a, h8v b) {
  return __builtin_shufflevector(a, b, 0,1,2,3,4,5,6,7,8,9,10,11,12,13,14,15);
}
static __device__ __forceinline__ h8v zero8() {
  h8v z = {0,0,0,0,0,0,0,0};
  return z;
}
// branch-free zeroing: AND the 4 dwords of an 8xf16 chunk with an all-ones/zero mask
static __device__ __forceinline__ h8v and_mask(h8v v, unsigned m) {
  u4v u = __builtin_bit_cast(u4v, v);
  u &= m;
  return __builtin_bit_cast(h8v, u);
}

// ---------------------------------------------------------------------------
// Weight packing into WMMA B-fragment layout.
// B (32x16 f16): lane L holds column n = L&15, elements j = K[(L>>4)*8 + (j&7) + (j>>3)*16].
// Packed layout: dst[ ((nt*KS + ks)*32 + lane)*16 + j ]
// ---------------------------------------------------------------------------
template<int CIN, int COUT>
__global__ void pack_conv_w(const float* __restrict__ src, _Float16* __restrict__ dst) {
  constexpr int K0 = 9 * CIN;
  constexpr int KP = (K0 + 31) & ~31;
  constexpr int KS = KP / 32;
  constexpr int NT = COUT / 16;
  int idx = blockIdx.x * blockDim.x + threadIdx.x;
  if (idx >= NT * KS * 512) return;
  int j    = idx & 15;
  int lane = (idx >> 4) & 31;
  int f    = idx >> 9;            // nt*KS + ks
  int ks   = f % KS;
  int nt   = f / KS;
  int k = ks * 32 + ((lane >> 4) << 3) + (j & 7) + ((j >> 3) << 4);
  int n = nt * 16 + (lane & 15);
  float v = 0.f;
  if (k < K0) {
    int s = k / CIN, c = k % CIN;          // s = ky*3+kx
    v = src[(n * CIN + c) * 9 + s];        // OIHW
  }
  dst[idx] = (_Float16)v;
}

__global__ void pack_proj_w(const float* __restrict__ w, _Float16* __restrict__ dst) {
  // K=128 (KS=4), N=128 (NT=8);  out[b,n] = sum_k mean[b,k] * w[n*128+k]
  int idx = blockIdx.x * blockDim.x + threadIdx.x;
  if (idx >= 8 * 4 * 512) return;
  int j    = idx & 15;
  int lane = (idx >> 4) & 31;
  int f    = idx >> 9;
  int ks   = f % 4;
  int nt   = f / 4;
  int k = ks * 32 + ((lane >> 4) << 3) + (j & 7) + ((j >> 3) << 4);
  int n = nt * 16 + (lane & 15);
  dst[idx] = (_Float16)w[n * 128 + k];
}

__global__ void pack_gru_w(const float* __restrict__ wih, const float* __restrict__ whh,
                           const float* __restrict__ bih, const float* __restrict__ bhh,
                           _Float16* __restrict__ dst) {
  // Effective W[K=320][N=512] over xcat = [xin(168) | h0(128) | 1 | 0-pad(23)]
  //  n in [0,256):   combined r,z pre-activation (gi+gh+biases), weight row n
  //  n in [256,384): gi_n (+b_ih_n), w_ih row n
  //  n in [384,512): gh_n (+b_hh_n), w_hh row n-128
  int idx = blockIdx.x * blockDim.x + threadIdx.x;
  if (idx >= 32 * 10 * 512) return;
  int j    = idx & 15;
  int lane = (idx >> 4) & 31;
  int f    = idx >> 9;
  int ks   = f % 10;
  int nt   = f / 10;
  int k = ks * 32 + ((lane >> 4) << 3) + (j & 7) + ((j >> 3) << 4);
  int n = nt * 16 + (lane & 15);
  float v = 0.f;
  if (n < 256) {
    if (k < 168)       v = wih[n * 168 + k];
    else if (k < 296)  v = whh[n * 128 + (k - 168)];
    else if (k == 296) v = bih[n] + bhh[n];
  } else if (n < 384) {
    if (k < 168)       v = wih[n * 168 + k];
    else if (k == 296) v = bih[n];
  } else {
    int r = n - 128;
    if (k >= 168 && k < 296) v = whh[r * 128 + (k - 168)];
    else if (k == 296)       v = bhh[r];
  }
  dst[idx] = (_Float16)v;
}

// ---------------------------------------------------------------------------
// conv1: Cin=1 (K=9 only -> not worth WMMA). Plain VALU, NHWC f16 out.
// ---------------------------------------------------------------------------
__global__ void conv1_k(const float* __restrict__ board, const float* __restrict__ w,
                        const float* __restrict__ bias, _Float16* __restrict__ out, int Bn) {
  int idx = blockIdx.x * blockDim.x + threadIdx.x;
  if (idx >= Bn * 1024) return;
  int x = idx & 31, y = (idx >> 5) & 31, b = idx >> 10;
  float p[9];
#pragma unroll
  for (int s = 0; s < 9; ++s) {
    int yy = y + s / 3 - 1, xx = x + s % 3 - 1;
    bool v = ((unsigned)yy < 32u) && ((unsigned)xx < 32u);
    p[s] = v ? board[((size_t)b << 10) + (yy << 5) + xx] : 0.f;
  }
#pragma unroll
  for (int o = 0; o < 16; ++o) {
    float a = bias[o];
#pragma unroll
    for (int s = 0; s < 9; ++s) a += p[s] * w[o * 9 + s];
    out[((size_t)idx << 4) + o] = (_Float16)fmaxf(a, 0.f);
  }
}

// ---------------------------------------------------------------------------
// 3x3 SAME conv as implicit GEMM with V_WMMA_F32_16X16X32_F16.
// Input NHWC f16 [B,H,W,CIN], output NHWC f16 [B,H,W,COUT], ReLU(conv+bias).
// Packed B weights are staged once per block into LDS (<=144KB, within the
// 320KB/WGP CDNA5 budget); the hot loop reads B via ds_load_b128 at
// lane*32 + constexpr offsets so the VMEM pipe carries only A halo loads.
// Each wave computes one 16-row M-tile x ALL output channels (NT acc tiles),
// grid-strided over M-tiles to amortize the staging.
// ---------------------------------------------------------------------------
template<int CIN, int COUT, int H, int W>
__global__ void conv3x3_wmma(const _Float16* __restrict__ in, const _Float16* __restrict__ wpk,
                             const float* __restrict__ bias, _Float16* __restrict__ out, int Bn) {
  constexpr int K0 = 9 * CIN;
  constexpr int KP = (K0 + 31) & ~31;
  constexpr int KS = KP / 32;
  constexpr int NT = COUT / 16;
  constexpr int BCHUNKS = NT * KS * 512 / 8;   // 16B chunks of packed B

  extern __shared__ _Float16 ldsb[];
  for (int i = threadIdx.x; i < BCHUNKS; i += blockDim.x)
    ((h8v*)ldsb)[i] = ((const h8v*)wpk)[i];
  __syncthreads();

  const int lane = threadIdx.x & 31;
  const int row = lane & 15, half = lane >> 4;
  const int Mt = Bn * H * W / 16;
  const int nwaves = gridDim.x * (blockDim.x >> 5);

  for (int tile = blockIdx.x * (blockDim.x >> 5) + (threadIdx.x >> 5); tile < Mt; tile += nwaves) {
    const int m = tile * 16 + row;
    const int b = m / (H * W);
    const int yx = m % (H * W);
    const int y = yx / W, x = yx % W;

    // per-shift clamped base offsets (always-valid address) + zero masks
    unsigned off[9];
    unsigned msk[9];
#pragma unroll
    for (int s = 0; s < 9; ++s) {
      int yy = y + s / 3 - 1, xx = x + s % 3 - 1;
      bool v = ((unsigned)yy < (unsigned)H) && ((unsigned)xx < (unsigned)W);
      int yc = yy < 0 ? 0 : (yy >= H ? H - 1 : yy);
      int xc = xx < 0 ? 0 : (xx >= W ? W - 1 : xx);
      off[s] = (unsigned)(((b * H + yc) * W + xc) * CIN + half * 8);
      msk[s] = v ? 0xFFFFFFFFu : 0u;
    }

    f8v acc[NT];
#pragma unroll
    for (int t = 0; t < NT; ++t) acc[t] = f8v{0.f,0.f,0.f,0.f,0.f,0.f,0.f,0.f};

#pragma unroll
    for (int ks = 0; ks < KS; ++ks) {
      h8v ac[2];
#pragma unroll
      for (int c2 = 0; c2 < 2; ++c2) {
        const int kb = ks * 32 + c2 * 16;      // constexpr after unroll
        if (kb < K0) {
          const int s  = kb / CIN;             // constexpr
          const int c0 = kb % CIN;             // constexpr
          h8v v = *(const h8v*)(in + (size_t)off[s] + c0);
          ac[c2] = and_mask(v, msk[s]);
        } else {
          ac[c2] = zero8();
        }
      }
      h16 Af = cat16(ac[0], ac[1]);
#pragma unroll
      for (int t = 0; t < NT; ++t) {
        const h8v* bp = (const h8v*)(ldsb + (size_t)((t * KS + ks) * 32 + lane) * 16);
        h16 Bf = cat16(bp[0], bp[1]);
        acc[t] = __builtin_amdgcn_wmma_f32_16x16x32_f16(false, Af, false, Bf, (short)0, acc[t], false, false);
      }
    }
    // D layout: VGPR v, lane L -> (M = v + 8*(L>>4), N = L&15)
    const int mb = tile * 16 + half * 8;
#pragma unroll
    for (int t = 0; t < NT; ++t) {
      const int ncol = t * 16 + row;
      const float bv = bias[ncol];
#pragma unroll
      for (int v = 0; v < 8; ++v) {
        float c = fmaxf(acc[t][v] + bv, 0.f);
        out[(size_t)(mb + v) * COUT + ncol] = (_Float16)c;
      }
    }
  }
}

// ---------------------------------------------------------------------------
// Generic WMMA GEMM: C[M,N] = A[M,LDA] * Wpk (+bias). All shapes constexpr,
// NB n-tiles per wave (A fragment reused NB times), fully unrolled.
// (Tiny fraction of total FLOPs; cached-global B is sufficient here.)
// ---------------------------------------------------------------------------
template<int NT, int KS, int NB, int LDA, int LDC, bool OUTF16>
__global__ void gemm_wmma(const _Float16* __restrict__ A, const _Float16* __restrict__ wpk,
                          const float* __restrict__ bias, void* __restrict__ outp, int M) {
  constexpr int NG = NT / NB;
  const int lane = threadIdx.x & 31;
  const int wave = blockIdx.x * (blockDim.x >> 5) + (threadIdx.x >> 5);
  if (wave >= (M >> 4) * NG) return;
  const int ng = wave % NG;
  const int mt = wave / NG;
  const int row = lane & 15, half = lane >> 4;
  const _Float16* arow = A + (size_t)(mt * 16 + row) * LDA;

  f8v acc[NB];
#pragma unroll
  for (int t = 0; t < NB; ++t) acc[t] = f8v{0.f,0.f,0.f,0.f,0.f,0.f,0.f,0.f};

#pragma unroll
  for (int ks = 0; ks < KS; ++ks) {
    h8v a0 = *(const h8v*)(arow + ks * 32 + half * 8);
    h8v a1 = *(const h8v*)(arow + ks * 32 + 16 + half * 8);
    h16 Af = cat16(a0, a1);
#pragma unroll
    for (int t = 0; t < NB; ++t) {
      const int nt = ng * NB + t;
      const h8v* bp = (const h8v*)(wpk + (size_t)((nt * KS + ks) * 32 + lane) * 16);
      h16 Bf = cat16(bp[0], bp[1]);
      acc[t] = __builtin_amdgcn_wmma_f32_16x16x32_f16(false, Af, false, Bf, (short)0, acc[t], false, false);
    }
  }
  const int mb = mt * 16 + half * 8;
#pragma unroll
  for (int t = 0; t < NB; ++t) {
    const int ncol = (ng * NB + t) * 16 + row;
    const float bv = bias ? bias[ncol] : 0.f;
#pragma unroll
    for (int v = 0; v < 8; ++v) {
      float c = acc[t][v] + bv;
      if (OUTF16) ((_Float16*)outp)[(size_t)(mb + v) * LDC + ncol] = (_Float16)c;
      else        ((float*)outp)[(size_t)(mb + v) * LDC + ncol] = c;
    }
  }
}

// ---------------------------------------------------------------------------
// 2x2 max pool (NHWC f16), vectorized 8 channels / thread
// ---------------------------------------------------------------------------
__global__ void maxpool2_k(const _Float16* __restrict__ in, _Float16* __restrict__ out,
                           int Bn, int H, int W, int C) {
  int idx = blockIdx.x * blockDim.x + threadIdx.x;
  int Ho = H >> 1, Wo = W >> 1, C8 = C >> 3;
  if (idx >= Bn * Ho * Wo * C8) return;
  int cg = idx % C8;
  int t = idx / C8;
  int x = t % Wo; t /= Wo;
  int y = t % Ho; int b = t / Ho;
  size_t base = ((size_t)(b * H + 2 * y) * W + 2 * x) * C + cg * 8;
  h8v v0 = *(const h8v*)(in + base);
  h8v v1 = *(const h8v*)(in + base + C);
  h8v v2 = *(const h8v*)(in + base + (size_t)W * C);
  h8v v3 = *(const h8v*)(in + base + (size_t)W * C + C);
  h8v o;
#pragma unroll
  for (int j = 0; j < 8; ++j)
    o[j] = (_Float16)fmaxf(fmaxf((float)v0[j], (float)v1[j]), fmaxf((float)v2[j], (float)v3[j]));
  *(h8v*)(out + ((size_t)(b * Ho + y) * Wo + x) * C + cg * 8) = o;
}

// global average over 8x8 spatial -> meanbuf f16 [B,128], 8 channels / thread
__global__ void mean_k(const _Float16* __restrict__ in, _Float16* __restrict__ out, int Bn) {
  int idx = blockIdx.x * blockDim.x + threadIdx.x;
  if (idx >= Bn * 16) return;
  int b = idx / 16, cg = idx % 16;
  const h8v* p = (const h8v*)(in + (size_t)b * 64 * 128 + cg * 8);
  float s[8] = {0.f,0.f,0.f,0.f,0.f,0.f,0.f,0.f};
#pragma unroll 8
  for (int i = 0; i < 64; ++i) {
    h8v v = p[i * 16];
#pragma unroll
    for (int j = 0; j < 8; ++j) s[j] += (float)v[j];
  }
  h8v o;
#pragma unroll
  for (int j = 0; j < 8; ++j) o[j] = (_Float16)(s[j] * (1.f / 64.f));
  *(h8v*)(out + (size_t)b * 128 + cg * 8) = o;
}

// fill xcat cols 128..319: action bits, user embedding, h0, constant-1, zero pad
__global__ void xcat_fill(const int* __restrict__ token, const int* __restrict__ uidx,
                          const float* __restrict__ emb, const float* __restrict__ h0,
                          _Float16* __restrict__ xcat, int Bn) {
  int idx = blockIdx.x * blockDim.x + threadIdx.x;
  if (idx >= Bn * 192) return;
  int b = idx / 192;
  int col = 128 + idx % 192;
  float v;
  if (col < 136)       v = (float)((token[b] >> (col - 128)) & 1);
  else if (col < 168)  v = emb[(size_t)uidx[b] * 32 + (col - 136)];
  else if (col < 296)  v = h0[(size_t)b * 128 + (col - 168)];
  else if (col == 296) v = 1.f;
  else                 v = 0.f;
  xcat[(size_t)b * 320 + col] = (_Float16)v;
}

// GRU gate math: g[B,512] = [r_pre | z_pre | gi_n | gh_n]  ->  h_new (f32)
__global__ void gates_k(const float* __restrict__ g, const float* __restrict__ h0,
                        float* __restrict__ hnew, int Bn) {
  int idx = blockIdx.x * blockDim.x + threadIdx.x;
  if (idx >= Bn * 128) return;
  int b = idx / 128, hh = idx % 128;
  const float* gb = g + (size_t)b * 512;
  float r = 1.f / (1.f + expf(-gb[hh]));
  float z = 1.f / (1.f + expf(-gb[128 + hh]));
  float n_ = tanhf(gb[256 + hh] + r * gb[384 + hh]);
  hnew[idx] = (1.f - z) * n_ + z * h0[idx];
}

// logits = h_new @ head_w.T + head_b  (B x 6, tiny GEMV)
__global__ void head_k(const float* __restrict__ hnew, const float* __restrict__ hw,
                       const float* __restrict__ hb, float* __restrict__ logits, int Bn) {
  int idx = blockIdx.x * blockDim.x + threadIdx.x;
  if (idx >= Bn * 6) return;
  int b = idx / 6, j = idx % 6;
  const float* hp = hnew + (size_t)b * 128;
  const float* wp = hw + j * 128;
  float a = hb[j];
#pragma unroll 8
  for (int c = 0; c < 128; ++c) a += hp[c] * wp[c];
  logits[idx] = a;
}

// ---------------------------------------------------------------------------

extern "C" void kernel_launch(void* const* d_in, const int* in_sizes, int n_in,
                              void* d_out, int out_size, void* d_ws, size_t ws_size,
                              hipStream_t stream) {
  const float* board = (const float*)d_in[0];
  const float* h0    = (const float*)d_in[1];
  const int*   token = (const int*)d_in[2];
  const int*   uidx  = (const int*)d_in[3];
  const float* c1w = (const float*)d_in[4],  *c1b = (const float*)d_in[5];
  const float* c2w = (const float*)d_in[6],  *c2b = (const float*)d_in[7];
  const float* c3w = (const float*)d_in[8],  *c3b = (const float*)d_in[9];
  const float* c4w = (const float*)d_in[10], *c4b = (const float*)d_in[11];
  const float* pw  = (const float*)d_in[12], *pb  = (const float*)d_in[13];
  const float* emb = (const float*)d_in[14];
  const float* wih = (const float*)d_in[15], *whh = (const float*)d_in[16];
  const float* bih = (const float*)d_in[17], *bhh = (const float*)d_in[18];
  const float* hw  = (const float*)d_in[19], *hb  = (const float*)d_in[20];
  float* out = (float*)d_out;
  const int Bn = in_sizes[0] / 1024;   // 4096

  // ---- workspace carve (regions reused across pipeline stages) ----
  char* ws = (char*)d_ws;
  size_t off = 0;
  auto carve = [&](size_t bytes) { size_t r = off; off = (off + bytes + 255) & ~(size_t)255; return r; };
  size_t oR0   = carve((size_t)Bn * 1024 * 32 * 2);  // act2 / act3 / act4
  size_t oR1   = carve((size_t)Bn * 1024 * 16 * 2);  // act1 / pooled2 / pooled3
  size_t oW2   = carve(32 * 160 * 2);
  size_t oW3   = carve(64 * 288 * 2);
  size_t oW4   = carve((size_t)128 * 576 * 2);
  size_t oWP   = carve((size_t)128 * 128 * 2);
  size_t oWG   = carve((size_t)512 * 320 * 2);
  size_t oMean = carve((size_t)Bn * 128 * 2);
  size_t oXcat = carve((size_t)Bn * 320 * 2);
  size_t oG    = carve((size_t)Bn * 512 * 4);
  (void)ws_size; (void)n_in; (void)out_size;

  _Float16* act1  = (_Float16*)(ws + oR1);
  _Float16* act2  = (_Float16*)(ws + oR0);
  _Float16* pool2 = (_Float16*)(ws + oR1);
  _Float16* act3  = (_Float16*)(ws + oR0);
  _Float16* pool3 = (_Float16*)(ws + oR1);
  _Float16* act4  = (_Float16*)(ws + oR0);
  _Float16* wpk2  = (_Float16*)(ws + oW2);
  _Float16* wpk3  = (_Float16*)(ws + oW3);
  _Float16* wpk4  = (_Float16*)(ws + oW4);
  _Float16* ppk   = (_Float16*)(ws + oWP);
  _Float16* gpk   = (_Float16*)(ws + oWG);
  _Float16* meanb = (_Float16*)(ws + oMean);
  _Float16* xcat  = (_Float16*)(ws + oXcat);
  float*    gbuf  = (float*)(ws + oG);
  float*    hnew  = out + (size_t)Bn * 6;

  auto cdiv = [](long long a, long long b) { return (unsigned)((a + b - 1) / b); };

  // ---- weight packing (device-side, cheap, deterministic) ----
  pack_conv_w<16, 32><<<cdiv(2 * 5 * 512, 256), 256, 0, stream>>>(c2w, wpk2);
  pack_conv_w<32, 64><<<cdiv(4 * 9 * 512, 256), 256, 0, stream>>>(c3w, wpk3);
  pack_conv_w<64, 128><<<cdiv(8 * 18 * 512, 256), 256, 0, stream>>>(c4w, wpk4);
  pack_proj_w<<<cdiv(8 * 4 * 512, 256), 256, 0, stream>>>(pw, ppk);
  pack_gru_w<<<cdiv(32 * 10 * 512, 256), 256, 0, stream>>>(wih, whh, bih, bhh, gpk);
  xcat_fill<<<cdiv((long long)Bn * 192, 256), 256, 0, stream>>>(token, uidx, emb, h0, xcat, Bn);

  // ---- CNN ----
  conv1_k<<<cdiv((long long)Bn * 1024, 256), 256, 0, stream>>>(board, c1w, c1b, act1, Bn);

  // grid-stride: ~4 M-tiles per wave, LDS-staged packed B per block
  {
    long long mt2 = (long long)Bn * 1024 / 16;
    conv3x3_wmma<16, 32, 32, 32><<<cdiv(mt2, 32), 256, 2 * 5 * 512 * 2, stream>>>(act1, wpk2, c2b, act2, Bn);
  }
  maxpool2_k<<<cdiv((long long)Bn * 256 * 4, 256), 256, 0, stream>>>(act2, pool2, Bn, 32, 32, 32);

  {
    long long mt3 = (long long)Bn * 256 / 16;
    conv3x3_wmma<32, 64, 16, 16><<<cdiv(mt3, 32), 256, 4 * 9 * 512 * 2, stream>>>(pool2, wpk3, c3b, act3, Bn);
  }
  maxpool2_k<<<cdiv((long long)Bn * 64 * 8, 256), 256, 0, stream>>>(act3, pool3, Bn, 16, 16, 64);

  {
    long long mt4 = (long long)Bn * 64 / 16;
    conv3x3_wmma<64, 128, 8, 8><<<cdiv(mt4, 32), 256, 8 * 18 * 512 * 2, stream>>>(pool3, wpk4, c4b, act4, Bn);
  }

  mean_k<<<cdiv((long long)Bn * 16, 256), 256, 0, stream>>>(act4, meanb, Bn);

  // proj: [B,128] x [128,128] + bias -> xcat cols 0..127 (f16, ldc=320)
  gemm_wmma<8, 4, 8, 128, 320, true><<<cdiv((long long)(Bn / 16) * 1, 8), 256, 0, stream>>>(meanb, ppk, pb, (void*)xcat, Bn);

  // GRU pre-activations: [B,320] x [320,512] -> g (f32, biases folded via 1-col)
  gemm_wmma<32, 10, 8, 320, 512, false><<<cdiv((long long)(Bn / 16) * 4, 8), 256, 0, stream>>>(xcat, gpk, nullptr, (void*)gbuf, Bn);

  gates_k<<<cdiv((long long)Bn * 128, 256), 256, 0, stream>>>(gbuf, h0, hnew, Bn);
  head_k<<<cdiv((long long)Bn * 6, 256), 256, 0, stream>>>(hnew, hw, hb, out, Bn);
}
